// LSTMTagger_9509057593680
// MI455X (gfx1250) — matrix operand order repaired and checked by reading, not compiled, not run on hardware
//
#include <hip/hip_runtime.h>
#include <hip/hip_bf16.h>
#include <stdint.h>

// Problem constants (match reference)
#define B_   64
#define S_   128
#define WP_  18
#define K_   3
#define E_   128
#define H_   256
#define L_   4
#define T_   50
#define W2H_ 512   // 2*H

typedef __attribute__((ext_vector_type(16))) __bf16 v16bf;
typedef __attribute__((ext_vector_type(2)))  __bf16 v2bf;
typedef __attribute__((ext_vector_type(8)))  float  v8f;
typedef __attribute__((ext_vector_type(2)))  float  v2f;

union Frag {
    uint32_t u[8];
    uint4    q[2];
    v16bf    v;
};

// Native packed f32 -> bf16 (lowers to v_cvt_pk_bf16_f32 on gfx1250)
__device__ __forceinline__ uint32_t pack2bf(float lo, float hi) {
    v2f f = {lo, hi};
    v2bf h = __builtin_convertvector(f, v2bf);
    return __builtin_bit_cast(uint32_t, h);
}
__device__ __forceinline__ uint16_t f2bf(float f) {
    __bf16 h = (__bf16)f;
    return __builtin_bit_cast(uint16_t, h);
}
__device__ __forceinline__ float bf2f(uint32_t hbits) {
    return __uint_as_float(hbits << 16);
}
__device__ __forceinline__ float sigmoidf_(float x) { return 1.0f / (1.0f + __expf(-x)); }

// ---------------------------------------------------------------------------
// Pack a row-major f32 weight W[Nreal][Kreal] into the bf16 WMMA B-operand
// layout: tile (nt,kt) of B[k][n] = W[n][k]; per tile 32 lanes x 8 dwords;
// lane l, reg r  ->  n = nt*16 + (l&15), k = kt*32 + (l>>4)*8 + (r&3)*2 + (r>>2)*16 + {0,1}.
// ---------------------------------------------------------------------------
__global__ void pack_b_kernel(const float* __restrict__ W, uint32_t* __restrict__ dst,
                              int NT, int KT, int Nreal, int Kreal) {
    int idx = blockIdx.x * blockDim.x + threadIdx.x;
    int total = NT * KT * 256;
    if (idx >= total) return;
    int r    = idx & 7;
    int lane = (idx >> 3) & 31;
    int tile = idx >> 8;
    int kt = tile % KT;
    int nt = tile / KT;
    int n = nt * 16 + (lane & 15);
    int k = kt * 32 + (lane >> 4) * 8 + (r & 3) * 2 + (r >> 2) * 16;
    float f0 = (n < Nreal && (k    ) < Kreal) ? W[(size_t)n * Kreal + k    ] : 0.f;
    float f1 = (n < Nreal && (k + 1) < Kreal) ? W[(size_t)n * Kreal + k + 1] : 0.f;
    dst[idx] = pack2bf(f0, f1);
}

// ---------------------------------------------------------------------------
// Gather word embeddings into time-major bf16 x[S][B][E].
// ---------------------------------------------------------------------------
__global__ void gather_x_kernel(const int* __restrict__ word_ids,
                                const float* __restrict__ word_emb,
                                uint16_t* __restrict__ xbf) {
    int idx = blockIdx.x * blockDim.x + threadIdx.x;   // S*B*E = 1048576
    if (idx >= S_ * B_ * E_) return;
    int e  = idx & (E_ - 1);
    int sb = idx >> 7;          // s*64 + b
    int b  = sb & (B_ - 1);
    int s  = sb >> 6;
    int wid = word_ids[b * S_ + s];
    xbf[idx] = f2bf(word_emb[(size_t)wid * E_ + e]);
}

// ---------------------------------------------------------------------------
// Char CNN + max-pool: one block (64 threads) per word. Thread = (pos, filt).
// ---------------------------------------------------------------------------
__global__ void char_cnn_kernel(const int* __restrict__ char_ids,
                                const float* __restrict__ char_emb,
                                const float* __restrict__ cnn_w,
                                const float* __restrict__ cnn_b,
                                float* __restrict__ out_char) {
    __shared__ float sh[64];
    int w   = blockIdx.x;          // 0 .. B*S-1 over [B][S]
    int tid = threadIdx.x;         // 64
    int filt = tid & 3;
    int pos  = tid >> 2;           // 0..15
    const int* cw = char_ids + (size_t)w * WP_;
    float acc = cnn_b[filt];
    for (int kk = 0; kk < K_; ++kk) {
        int cid = cw[pos + kk];
        const float* ce = char_emb + (size_t)cid * E_;
        const float* wt = cnn_w + (size_t)filt * (K_ * E_) + kk * E_;
        #pragma unroll 4
        for (int e = 0; e < E_; ++e) acc = fmaf(ce[e], wt[e], acc);
    }
    sh[tid] = acc;
    __syncthreads();
    if (tid < 4) {
        float m = sh[tid];
        #pragma unroll
        for (int p = 1; p < 16; ++p) m = fmaxf(m, sh[p * 4 + tid]);
        out_char[(size_t)w * L_ + tid] = m;
    }
}

// ---------------------------------------------------------------------------
// Bulk pre-activation GEMM: Zx = x @ Wih^T + bias for both directions.
// One wave per 16x16 output tile; result stored bf16 in C-fragment layout:
//   zx[((dir*512 + mtile)*64 + nt)*32 + lane][r(8 bf16)]  (16 bytes/lane/tile)
// so the recurrent kernel initializes accumulators with a single b128 load.
// ---------------------------------------------------------------------------
__global__ __launch_bounds__(256) void xw_kernel(
    const uint16_t* __restrict__ xbf,
    const uint32_t* __restrict__ wihf_pk, const float* __restrict__ b_f,
    const uint32_t* __restrict__ wihb_pk, const float* __restrict__ b_b,
    uint16_t* __restrict__ zx) {

    const int wid  = blockIdx.x * 8 + (threadIdx.x >> 5);   // 0 .. 65535
    const int lane = threadIdx.x & 31;
    const int h2   = lane >> 4;
    const int ncol = lane & 15;

    const int nt    = wid & 63;
    const int rest  = wid >> 6;
    const int mtile = rest & 511;
    const int dir   = rest >> 9;

    const uint32_t* wih  = dir ? wihb_pk : wihf_pk;
    const float*    bias = dir ? b_b     : b_f;

    const float bv = bias[nt * 16 + ncol];
    v8f acc = {bv, bv, bv, bv, bv, bv, bv, bv};

    const int M = mtile * 16 + ncol;
    for (int kt = 0; kt < 4; ++kt) {
        const int k0 = kt * 32 + h2 * 8;
        Frag a;
        const uint16_t* p = xbf + (size_t)M * E_ + k0;
        a.q[0] = *(const uint4*)(p);
        a.q[1] = *(const uint4*)(p + 16);
        Frag bm;
        const uint32_t* bp = wih + ((size_t)(nt * 4 + kt) * 32 + lane) * 8;
        bm.q[0] = *(const uint4*)(bp);
        bm.q[1] = *(const uint4*)(bp + 4);
        acc = __builtin_amdgcn_wmma_f32_16x16x32_bf16(
            false, a.v, false, bm.v, (short)0, acc, false, false);
    }

    uint4 out;
    out.x = pack2bf(acc[0], acc[1]);
    out.y = pack2bf(acc[2], acc[3]);
    out.z = pack2bf(acc[4], acc[5]);
    out.w = pack2bf(acc[6], acc[7]);
    *(uint4*)(zx + ((size_t)wid * 32 + lane) * 8) = out;
}

// ---------------------------------------------------------------------------
// Persistent BiLSTM recurrence: grid = 2 blocks (dir), block = 512 = 16 waves.
// Wave w owns columns j in [w*16, w*16+16) of H for all 4 gates.
// Per step: acc init from precomputed Zx (one b128 per tile), then only
// h_{t-1} @ Whh^T (8 K-iters of v_wmma_f32_16x16x32_bf16) + gate math.
// c state lives in VGPRs across all 128 steps; h lives in LDS as bf16.
// Only one Whh B-fragment is live at a time, and an asm barrier makes the
// Whh pointer opaque per-step so loop-invariant fragments are not hoisted.
// ---------------------------------------------------------------------------
__global__ __launch_bounds__(512) void lstm_kernel(
    const uint16_t* __restrict__ zx,
    const uint32_t* __restrict__ whhpk_f,
    const uint32_t* __restrict__ whhpk_b,
    float* __restrict__ hs_f, float* __restrict__ hs_b) {

    const int dir = blockIdx.x;
    const uint32_t* whh = dir ? whhpk_b : whhpk_f;
    float*          hs  = dir ? hs_b    : hs_f;

    const int tid  = threadIdx.x;
    const int wave = tid >> 5;          // 0..15
    const int lane = tid & 31;
    const int h2   = lane >> 4;
    const int ncol = lane & 15;
    const int j    = wave * 16 + ncol;

    constexpr int LDP = 264;            // padded row pitch (ushorts)
    __shared__ uint16_t lds_h[64 * LDP];

    for (int i = tid; i < 64 * LDP; i += 512) lds_h[i] = 0;
    __syncthreads();

    v8f c[4];
    #pragma unroll
    for (int mt = 0; mt < 4; ++mt) {
        v8f z = {0.f, 0.f, 0.f, 0.f, 0.f, 0.f, 0.f, 0.f};
        c[mt] = z;
    }

    uint64_t whh_u = (uint64_t)whh;

    for (int step = 0; step < S_; ++step) {
        const int t = dir ? (S_ - 1 - step) : step;

        // opaque per-step copy of the weight pointer: defeats LICM hoisting
        uint64_t whh_s = whh_u;
        asm volatile("" : "+v"(whh_s));
        const uint32_t* whh_t = (const uint32_t*)whh_s;

        // ---- init accumulators from Zx (x@Wih^T + bias, C-frag layout) ----
        v8f acc[4][4];
        #pragma unroll
        for (int g = 0; g < 4; ++g) {
            #pragma unroll
            for (int mt = 0; mt < 4; ++mt) {
                const size_t tile = ((size_t)(dir * 512 + t * 4 + mt) * 64
                                     + (g * 16 + wave));
                uint4 zq = *(const uint4*)(zx + (tile * 32 + lane) * 8);
                uint32_t zu[4] = {zq.x, zq.y, zq.z, zq.w};
                v8f a;
                #pragma unroll
                for (int r = 0; r < 8; ++r)
                    a[r] = bf2f((zu[r >> 1] >> ((r & 1) * 16)) & 0xFFFFu);
                acc[g][mt] = a;
            }
        }

        // ---- h_{t-1} @ Whh^T  (K = 256) ----
        for (int kt = 0; kt < 8; ++kt) {
            const int k0 = kt * 32 + h2 * 8;
            Frag a[4];
            #pragma unroll
            for (int mt = 0; mt < 4; ++mt) {
                const uint16_t* p = &lds_h[(mt * 16 + ncol) * LDP + k0];
                a[mt].q[0] = *(const uint4*)(p);
                a[mt].q[1] = *(const uint4*)(p + 16);
            }
            if (kt + 1 < 8) // gfx1250 global_prefetch_b8 for next fragment
                __builtin_prefetch(whh_t + ((size_t)(wave * 8 + kt + 1) * 32 + lane) * 8, 0, 3);
            #pragma unroll
            for (int g = 0; g < 4; ++g) {
                Frag bm;   // single live B-fragment
                const uint32_t* bp = whh_t + ((size_t)((g * 16 + wave) * 8 + kt) * 32 + lane) * 8;
                bm.q[0] = *(const uint4*)(bp);
                bm.q[1] = *(const uint4*)(bp + 4);
                #pragma unroll
                for (int mt = 0; mt < 4; ++mt)
                    acc[g][mt] = __builtin_amdgcn_wmma_f32_16x16x32_bf16(
                        false, a[mt].v, false, bm.v, (short)0, acc[g][mt], false, false);
            }
        }

        __syncthreads();   // all waves finished reading old h

        // ---- gate math in-register; write new h (global f32 + LDS bf16) ----
        #pragma unroll
        for (int mt = 0; mt < 4; ++mt) {
            #pragma unroll
            for (int r = 0; r < 8; ++r) {
                const int brow = mt * 16 + h2 * 8 + r;
                float zi = acc[0][mt][r];
                float zf = acc[1][mt][r];
                float zg = acc[2][mt][r];
                float zo = acc[3][mt][r];
                float cc = sigmoidf_(zf) * c[mt][r] + sigmoidf_(zi) * tanhf(zg);
                c[mt][r] = cc;
                float hv = sigmoidf_(zo) * tanhf(cc);
                hs[((size_t)t * 64 + brow) * H_ + j] = hv;
                lds_h[brow * LDP + j] = f2bf(hv);
            }
        }
        __syncthreads();   // new h visible to all waves
    }
}

// ---------------------------------------------------------------------------
// Output linear: tag[b][s][n] = [hf|hb](s,b,:) @ out_w^T + out_b  (WMMA, N pad 64)
// One wave per 16x16 tile; M = s*64+b; K = 512 (hf then hb).
// ---------------------------------------------------------------------------
__global__ __launch_bounds__(256) void linear_kernel(
    const float* __restrict__ hf, const float* __restrict__ hb,
    const uint32_t* __restrict__ owpk, const float* __restrict__ out_b,
    float* __restrict__ tag) {

    const int wid  = blockIdx.x * 8 + (threadIdx.x >> 5);   // 0..2047
    const int lane = threadIdx.x & 31;
    const int h2   = lane >> 4;
    const int ncol = lane & 15;
    const int nt    = wid & 3;
    const int mtile = wid >> 2;                             // 0..511
    const int n = nt * 16 + ncol;

    const float bvv = (n < T_) ? out_b[n] : 0.f;
    v8f acc = {bvv, bvv, bvv, bvv, bvv, bvv, bvv, bvv};

    const int M = mtile * 16 + ncol;                        // A-row for this lane
    for (int kt = 0; kt < 16; ++kt) {
        const float* hsrc = (kt < 8) ? hf : hb;
        const int k0 = (kt & 7) * 32 + h2 * 8;
        const float* p = hsrc + (size_t)M * H_ + k0;
        float f[16];
        #pragma unroll
        for (int i = 0; i < 8; ++i) f[i] = p[i];
        #pragma unroll
        for (int i = 0; i < 8; ++i) f[8 + i] = p[16 + i];
        Frag a;
        #pragma unroll
        for (int r = 0; r < 8; ++r)
            a.u[r] = pack2bf(f[2 * r], f[2 * r + 1]);
        Frag bm;
        const uint32_t* bp = owpk + ((size_t)(nt * 16 + kt) * 32 + lane) * 8;
        bm.q[0] = *(const uint4*)(bp);
        bm.q[1] = *(const uint4*)(bp + 4);
        acc = __builtin_amdgcn_wmma_f32_16x16x32_bf16(
            false, a.v, false, bm.v, (short)0, acc, false, false);
    }

    if (n < T_) {
        #pragma unroll
        for (int r = 0; r < 8; ++r) {
            int Mr = mtile * 16 + h2 * 8 + r;
            int s = Mr >> 6;
            int b = Mr & 63;
            tag[((size_t)b * S_ + s) * T_ + n] = acc[r];
        }
    }
}

// ---------------------------------------------------------------------------
// log_softmax over the sequence axis: one block per (b, t), 128 threads = s.
// ---------------------------------------------------------------------------
__global__ void softmax_kernel(const float* __restrict__ tag, float* __restrict__ out) {
    __shared__ float sh[128];
    int bt = blockIdx.x;            // 0 .. B*T-1
    int b = bt / T_;
    int t = bt % T_;
    int s = threadIdx.x;            // 0..127
    float v = tag[((size_t)b * S_ + s) * T_ + t];
    sh[s] = v;
    __syncthreads();
    for (int off = 64; off > 0; off >>= 1) {
        if (s < off) sh[s] = fmaxf(sh[s], sh[s + off]);
        __syncthreads();
    }
    float m = sh[0];
    __syncthreads();
    sh[s] = __expf(v - m);
    __syncthreads();
    for (int off = 64; off > 0; off >>= 1) {
        if (s < off) sh[s] += sh[s + off];
        __syncthreads();
    }
    float lse = m + __logf(sh[0]);
    out[((size_t)b * S_ + s) * T_ + t] = v - lse;
}

// ---------------------------------------------------------------------------
// Host-side launcher
// ---------------------------------------------------------------------------
extern "C" void kernel_launch(void* const* d_in, const int* in_sizes, int n_in,
                              void* d_out, int out_size, void* d_ws, size_t ws_size,
                              hipStream_t stream) {
    (void)in_sizes; (void)n_in; (void)out_size; (void)ws_size;

    const int*   char_ids = (const int*)d_in[0];
    const int*   word_ids = (const int*)d_in[1];
    const float* char_emb = (const float*)d_in[2];
    const float* word_emb = (const float*)d_in[3];
    const float* cnn_w    = (const float*)d_in[4];
    const float* cnn_b    = (const float*)d_in[5];
    const float* wih_f    = (const float*)d_in[6];
    const float* whh_f    = (const float*)d_in[7];
    const float* b_f      = (const float*)d_in[8];
    const float* wih_b    = (const float*)d_in[9];
    const float* whh_b    = (const float*)d_in[10];
    const float* b_b      = (const float*)d_in[11];
    const float* out_w    = (const float*)d_in[12];
    const float* out_b    = (const float*)d_in[13];

    float* out_tag  = (float*)d_out;                         // [B][S][T]
    float* out_char = (float*)d_out + (size_t)B_ * S_ * T_;  // [B][S][L]

    // workspace carve-up
    uint8_t* ws = (uint8_t*)d_ws;
    size_t off = 0;
    auto carve = [&](size_t bytes) -> void* {
        void* p = ws + off;
        off = (off + bytes + 255) & ~(size_t)255;
        return p;
    };
    uint16_t* x_bf    = (uint16_t*)carve((size_t)S_ * B_ * E_ * 2);      // 2 MB
    uint32_t* wihf_pk = (uint32_t*)carve((size_t)64 * 4  * 256 * 4);     // 256 KB
    uint32_t* wihb_pk = (uint32_t*)carve((size_t)64 * 4  * 256 * 4);
    uint32_t* whhf_pk = (uint32_t*)carve((size_t)64 * 8  * 256 * 4);     // 512 KB
    uint32_t* whhb_pk = (uint32_t*)carve((size_t)64 * 8  * 256 * 4);
    uint32_t* ow_pk   = (uint32_t*)carve((size_t)4  * 16 * 256 * 4);     // 64 KB
    uint16_t* zx      = (uint16_t*)carve((size_t)2 * 512 * 64 * 256 * 2);// 32 MB
    float*    hf      = (float*)carve((size_t)S_ * B_ * H_ * 4);         // 8 MB
    float*    hb      = (float*)carve((size_t)S_ * B_ * H_ * 4);         // 8 MB
    float*    tag_ws  = (float*)carve((size_t)B_ * S_ * T_ * 4);         // 1.6 MB

    // 1) pack weights into WMMA B-operand bf16 layout
    pack_b_kernel<<<(64 * 4 * 256 + 255) / 256, 256, 0, stream>>>(wih_f, wihf_pk, 64, 4, 1024, 128);
    pack_b_kernel<<<(64 * 4 * 256 + 255) / 256, 256, 0, stream>>>(wih_b, wihb_pk, 64, 4, 1024, 128);
    pack_b_kernel<<<(64 * 8 * 256 + 255) / 256, 256, 0, stream>>>(whh_f, whhf_pk, 64, 8, 1024, 256);
    pack_b_kernel<<<(64 * 8 * 256 + 255) / 256, 256, 0, stream>>>(whh_b, whhb_pk, 64, 8, 1024, 256);
    pack_b_kernel<<<(4 * 16 * 256 + 255) / 256, 256, 0, stream>>>(out_w, ow_pk, 4, 16, T_, W2H_);

    // 2) gather word embeddings (time-major, bf16)
    gather_x_kernel<<<(S_ * B_ * E_ + 255) / 256, 256, 0, stream>>>(word_ids, word_emb, x_bf);

    // 3) char CNN branch (independent output)
    char_cnn_kernel<<<B_ * S_, 64, 0, stream>>>(char_ids, char_emb, cnn_w, cnn_b, out_char);

    // 4) bulk pre-activation GEMM Zx = x@Wih^T + bias (both directions)
    xw_kernel<<<8192, 256, 0, stream>>>(x_bf, wihf_pk, b_f, wihb_pk, b_b, zx);

    // 5) persistent BiLSTM recurrence (2 blocks: fwd, bwd)
    lstm_kernel<<<2, 512, 0, stream>>>(zx, whhf_pk, whhb_pk, hf, hb);

    // 6) output linear (WMMA)
    linear_kernel<<<256, 256, 0, stream>>>(hf, hb, ow_pk, out_b, tag_ws);

    // 7) log_softmax over sequence axis
    softmax_kernel<<<B_ * T_, 128, 0, stream>>>(tag_ws, out_tag);
}
